// DecoderWithAttention_35038343201342
// MI455X (gfx1250) — matrix operand order repaired and checked by reading, not compiled, not run on hardware
//
#include <hip/hip_runtime.h>
#include <hip/hip_bf16.h>

// ---------------------------------------------------------------------------
// Types for CDNA5 WMMA (wave32): bf16 16x16x32 -> f32 accumulate
// ---------------------------------------------------------------------------
typedef __bf16 bf16_t;
typedef __attribute__((ext_vector_type(16))) __bf16 v16bf;
typedef __attribute__((ext_vector_type(8)))  float  v8f;

union frag16 { v16bf v; unsigned int u[8]; };

__device__ __forceinline__ bf16_t f32_to_bf16(float f) {
  unsigned int u = __builtin_bit_cast(unsigned int, f);
  unsigned int r = (u + 0x7FFFu + ((u >> 16) & 1u)) >> 16;  // RNE
  unsigned short s = (unsigned short)r;
  return __builtin_bit_cast(bf16_t, s);
}

__device__ __forceinline__ float sigm(float x) {
  return 1.0f / (1.0f + __expf(-x));
}

// Load one k-chunk of fragments: 1 B frag (16 cols x 32 K) + 4 A frags
// (4 stacked 16-row tiles x 32 K). 10 x b128 per lane group, all contiguous.
__device__ __forceinline__ void load_frags(const bf16_t* __restrict__ wrow,
                                           const bf16_t* __restrict__ abase,
                                           int lda, int k,
                                           frag16& ub, frag16 ua[4]) {
  const unsigned int* bp = (const unsigned int*)(wrow + k);
  #pragma unroll
  for (int i = 0; i < 8; ++i) ub.u[i] = bp[i];
  #pragma unroll
  for (int mt = 0; mt < 4; ++mt) {
    const bf16_t* arow = abase + (size_t)(mt * 16) * lda + k;
    const unsigned int* ap0 = (const unsigned int*)(arow);       // K=base+0..7
    const unsigned int* ap1 = (const unsigned int*)(arow + 8);   // K=base+8..15
    #pragma unroll
    for (int i = 0; i < 4; ++i) { ua[mt].u[i] = ap0[i]; ua[mt].u[4 + i] = ap1[i]; }
  }
}

__device__ __forceinline__ void wmma_group(const frag16 ua[4], const frag16& ub,
                                           v8f acc[4]) {
  #pragma unroll
  for (int mt = 0; mt < 4; ++mt)
    acc[mt] = __builtin_amdgcn_wmma_f32_16x16x32_bf16(
        false, ua[mt].v, false, ub.v, (short)0, acc[mt], false, false);
}

// ---------------------------------------------------------------------------
// Column-strip WMMA: one wave computes a full 64-row x 16-col strip
// (4 stacked 16x16 tiles). Software-pipelined 2 deep: the WMMAs for chunk k
// run while the loads for chunk k+64 are in flight, so waits never drain to
// zero. B fragment is loaded once per k-chunk and reused by 4 WMMAs; A comes
// from the tiny WGP$-resident activation matrix -> minimal L2 weight traffic.
// Per-lane element mapping used here (A and B share contiguous k; C/D as in ISA 7.12.2):
//   A (16-bit 16x32): lane = half*16 + m;  K = half*16 + {0..15}
//   B (16-bit 32x16): lane = half*16 + n;  K = half*16 + {0..15}
//   C/D (f32 16x16):  lane = half*16 + n;  vgpr r -> M = half*8 + r
// Requires K % 64 == 0 and K >= 64.
// ---------------------------------------------------------------------------
__device__ __forceinline__ void wmma_colstrip_bf16(
    const bf16_t* __restrict__ A, int lda,   // [M,K] activations, rows mg0..mg0+63
    const bf16_t* __restrict__ W, int ldw,   // [N,K] weights
    int mg0, int n0, int K, v8f acc[4]) {
  const int lane = threadIdx.x & 31;
  const int half = lane >> 4;     // 0 or 1
  const int l16  = lane & 15;
  const bf16_t* wrow  = W + (size_t)(n0 + l16) * ldw + half * 16;
  const bf16_t* abase = A + (size_t)(mg0 + l16) * lda + half * 16;
  frag16 ub0, ua0[4], ub1, ua1[4];
  load_frags(wrow, abase, lda, 0,  ub0, ua0);
  load_frags(wrow, abase, lda, 32, ub1, ua1);
  for (int k = 0; k < K - 64; k += 64) {
    wmma_group(ua0, ub0, acc);
    load_frags(wrow, abase, lda, k + 64, ub0, ua0);
    wmma_group(ua1, ub1, acc);
    load_frags(wrow, abase, lda, k + 96, ub1, ua1);
  }
  wmma_group(ua0, ub0, acc);
  wmma_group(ua1, ub1, acc);
}

// Generic D[M,N] = A[M,K](bf16) @ W[N,K]^T(bf16) + bias, D f32 row-major ldd.
// Requires M % 64 == 0, N % 16 == 0, K % 64 == 0.
__global__ __launch_bounds__(256)
void gemm_bias_kernel(const bf16_t* __restrict__ A, int lda,
                      const bf16_t* __restrict__ W, int ldw,
                      const float* __restrict__ bias,
                      float* __restrict__ D, int ldd,
                      int M, int N, int K) {
  const int wavesPerBlock = blockDim.x >> 5;
  const int wave = blockIdx.x * wavesPerBlock + (threadIdx.x >> 5);
  const int tilesN = N >> 4;
  const int mgroups = M >> 6;
  if (wave >= mgroups * tilesN) return;          // wave-uniform: EXEC stays full
  const int mg0 = (wave / tilesN) << 6;
  const int n0 = (wave % tilesN) << 4;
  v8f acc[4] = {{}, {}, {}, {}};
  wmma_colstrip_bf16(A, lda, W, ldw, mg0, n0, K, acc);
  const int lane = threadIdx.x & 31;
  const int ncol = n0 + (lane & 15);
  const float bv = bias[ncol];
  #pragma unroll
  for (int mt = 0; mt < 4; ++mt) {
    const int mbase = mg0 + mt * 16 + ((lane >> 4) << 3);
    #pragma unroll
    for (int r = 0; r < 8; ++r)
      D[(size_t)(mbase + r) * ldd + ncol] = acc[mt][r] + bv;
  }
}

// One-time att1 GEMM with TRANSPOSED store: att1T[b][c][p] so the per-step
// attention kernel reads fully coalesced P-contiguous rows.
// A = encPB [16384,512], W = Wenc16 [512,512]; m = b*256+p, n = c.
__global__ __launch_bounds__(256)
void gemm_att1_kernel(const bf16_t* __restrict__ A,
                      const bf16_t* __restrict__ W,
                      const float* __restrict__ bias,
                      float* __restrict__ att1T) {   // [B,512,256]
  const int wavesPerBlock = blockDim.x >> 5;
  const int wave = blockIdx.x * wavesPerBlock + (threadIdx.x >> 5);
  const int tilesN = 512 >> 4;                   // 32
  if (wave >= 256 * tilesN) return;              // 16384/64 mgroups * 32
  const int mg0 = (wave / tilesN) << 6;
  const int n0 = (wave % tilesN) << 4;
  v8f acc[4] = {{}, {}, {}, {}};
  wmma_colstrip_bf16(A, 512, W, 512, mg0, n0, 512, acc);
  const int lane = threadIdx.x & 31;
  const int c = n0 + (lane & 15);
  const float bv = bias[c];
  #pragma unroll
  for (int mt = 0; mt < 4; ++mt) {
    const int mbase = mg0 + mt * 16 + ((lane >> 4) << 3);
    #pragma unroll
    for (int r = 0; r < 8; ++r) {
      const int m = mbase + r;                   // b*256 + p
      const int b = m >> 8, p = m & 255;
      att1T[((size_t)b * 512 + c) * 256 + p] = acc[mt][r] + bv;
    }
  }
}

// Vocab projection with per-(b,t) masking, writes out[b][t][v]. M = 64 (batch).
__global__ __launch_bounds__(256)
void gemm_pred_kernel(const bf16_t* __restrict__ A, int lda,
                      const bf16_t* __restrict__ Wfc,
                      const float* __restrict__ bfc,
                      const int* __restrict__ lens,
                      float* __restrict__ out,
                      int t, int T, int V, int K) {
  const int wavesPerBlock = blockDim.x >> 5;
  const int wave = blockIdx.x * wavesPerBlock + (threadIdx.x >> 5);
  const int tilesN = V >> 4;
  if (wave >= tilesN) return;                    // single 64-row group
  const int n0 = wave << 4;
  v8f acc[4] = {{}, {}, {}, {}};
  wmma_colstrip_bf16(A, lda, Wfc, K, 0, n0, K, acc);
  const int lane = threadIdx.x & 31;
  const int ncol = n0 + (lane & 15);
  const float bv = bfc[ncol];
  #pragma unroll
  for (int mt = 0; mt < 4; ++mt) {
    const int mbase = mt * 16 + ((lane >> 4) << 3);
    #pragma unroll
    for (int r = 0; r < 8; ++r) {
      const int b = mbase + r;
      const bool active = t < (lens[b] - 1);
      out[((size_t)b * T + t) * V + ncol] = active ? (acc[mt][r] + bv) : 0.0f;
    }
  }
}

// ---------------------------------------------------------------------------
// Attention step (one block per batch row, 256 threads == P):
//   e_p = b_full + w_full . relu(att1T[b,:,p] + att2[b,:]); softmax over p;
//   awe_c = sigmoid(gate_pre_c) * sum_p alpha_p * enc[b,c,p];
//   xinh[b] = [ emb_t(256) | awe(512) | h(512, written elsewhere) ] in bf16
// Both big loops stream P-contiguous rows -> fully coalesced.
// ---------------------------------------------------------------------------
__global__ __launch_bounds__(256)
void attention_kernel(const float* __restrict__ att1T,    // [B,512,256]
                      const float* __restrict__ attg,     // [B,1024] = att2 | gate_pre
                      const float* __restrict__ encBCP,   // encoder_out [B,512,256]
                      const float* __restrict__ w_full,   // [512]
                      const float* __restrict__ b_full,   // [1]
                      const bf16_t* __restrict__ embT,    // [B,256] (this t)
                      bf16_t* __restrict__ xinh) {        // [B,1280]
  __shared__ float s_att2[512];
  __shared__ float s_w[512];
  __shared__ float s_e[256];
  __shared__ float s_red[256];
  const int b = blockIdx.x;
  const int tid = threadIdx.x;
  for (int i = tid; i < 512; i += 256) {
    s_att2[i] = attg[b * 1024 + i];
    s_w[i] = w_full[i];
  }
  __syncthreads();
  {
    const float* acol = att1T + (size_t)b * 512 * 256 + tid;
    float e = b_full[0];
    #pragma unroll 4
    for (int c = 0; c < 512; ++c) {
      float v = acol[(size_t)c * 256] + s_att2[c];
      v = v > 0.0f ? v : 0.0f;
      e = fmaf(s_w[c], v, e);
    }
    s_e[tid] = e;
  }
  __syncthreads();
  // softmax over 256 positions
  s_red[tid] = s_e[tid];
  __syncthreads();
  for (int s = 128; s > 0; s >>= 1) {
    if (tid < s) s_red[tid] = fmaxf(s_red[tid], s_red[tid + s]);
    __syncthreads();
  }
  const float mx = s_red[0];
  __syncthreads();
  const float ex = __expf(s_e[tid] - mx);
  s_e[tid] = ex;
  s_red[tid] = ex;
  __syncthreads();
  for (int s = 128; s > 0; s >>= 1) {
    if (tid < s) s_red[tid] += s_red[tid + s];
    __syncthreads();
  }
  const float inv = 1.0f / s_red[0];
  __syncthreads();
  s_e[tid] *= inv;           // alpha
  __syncthreads();
  // gated attention-weighted encoding; enc read contiguous in p
  for (int c = tid; c < 512; c += 256) {
    const float* erow = encBCP + ((size_t)b * 512 + c) * 256;
    float s = 0.0f;
    #pragma unroll 4
    for (int p = 0; p < 256; ++p) s = fmaf(s_e[p], erow[p], s);
    const float g = sigm(attg[b * 1024 + 512 + c]);
    xinh[(size_t)b * 1280 + 256 + c] = f32_to_bf16(s * g);
  }
  xinh[(size_t)b * 1280 + tid] = embT[b * 256 + tid];
}

// LSTM cell elementwise: consumes gates [B,2048] (i|f|g|o), updates c (f32)
// and writes bf16 h into the xinh tail so the next GEMMs read it directly.
__global__ __launch_bounds__(256)
void lstm_kernel(const float* __restrict__ gates,
                 float* __restrict__ cstate,
                 bf16_t* __restrict__ xinh) {
  const int idx = blockIdx.x * blockDim.x + threadIdx.x;
  if (idx >= 64 * 512) return;
  const int b = idx >> 9, d = idx & 511;
  const float* g = gates + (size_t)b * 2048;
  const float si = sigm(g[d]);
  const float sf = sigm(g[512 + d]);
  const float gg = tanhf(g[1024 + d]);
  const float so = sigm(g[1536 + d]);
  const float c2 = sf * cstate[idx] + si * gg;
  cstate[idx] = c2;
  xinh[(size_t)b * 1280 + 768 + d] = f32_to_bf16(so * tanhf(c2));
}

// ---------------------------- one-time setup kernels -----------------------
__global__ void cvt_kernel(bf16_t* __restrict__ dst, const float* __restrict__ src, int n) {
  for (int i = blockIdx.x * blockDim.x + threadIdx.x; i < n; i += gridDim.x * blockDim.x)
    dst[i] = f32_to_bf16(src[i]);
}

__global__ void wcomb_kernel(bf16_t* __restrict__ Wcomb,
                             const float* __restrict__ Wih,
                             const float* __restrict__ Whh, int n) {  // n = 2048*1280
  for (int i = blockIdx.x * blockDim.x + threadIdx.x; i < n; i += gridDim.x * blockDim.x) {
    const int row = i / 1280, k = i - row * 1280;
    const float v = (k < 768) ? Wih[(size_t)row * 768 + k] : Whh[(size_t)row * 512 + (k - 768)];
    Wcomb[i] = f32_to_bf16(v);
  }
}

__global__ void bcomb_kernel(float* __restrict__ bcomb,
                             const float* __restrict__ bih,
                             const float* __restrict__ bhh) {
  const int i = blockIdx.x * blockDim.x + threadIdx.x;
  if (i < 2048) bcomb[i] = bih[i] + bhh[i];
}

__global__ void batt_kernel(float* __restrict__ batt,
                            const float* __restrict__ bdec,
                            const float* __restrict__ bbeta) {
  const int i = blockIdx.x * blockDim.x + threadIdx.x;
  if (i < 1024) batt[i] = (i < 512) ? bdec[i] : bbeta[i - 512];
}

__global__ void embed_kernel(const float* __restrict__ emb,
                             const int* __restrict__ caps,
                             bf16_t* __restrict__ embTB) {  // [T=31][B=64][256]
  const int t = blockIdx.x >> 6;
  const int b = blockIdx.x & 63;
  const int tok = caps[b * 32 + t];
  embTB[((size_t)(t * 64 + b)) * 256 + threadIdx.x] =
      f32_to_bf16(emb[(size_t)tok * 256 + threadIdx.x]);
}

__global__ void enc_bf16_kernel(const float* __restrict__ encBCP,   // [B,512,256]
                                bf16_t* __restrict__ encPB) {       // [B*256,512]
  const int n = 64 * 256 * 512;
  for (int i = blockIdx.x * blockDim.x + threadIdx.x; i < n; i += gridDim.x * blockDim.x) {
    const int c = i & 511; const int rest = i >> 9;
    const int p = rest & 255; const int b = rest >> 8;
    encPB[i] = f32_to_bf16(encBCP[((size_t)b * 512 + c) * 256 + p]);
  }
}

__global__ void mean_kernel(const float* __restrict__ encBCP, float* __restrict__ meanenc) {
  const int idx = blockIdx.x * blockDim.x + threadIdx.x;  // b*512 + c
  if (idx >= 64 * 512) return;
  const float* row = encBCP + (size_t)idx * 256;
  float s = 0.0f;
  #pragma unroll 4
  for (int p = 0; p < 256; ++p) s += row[p];
  meanenc[idx] = s * (1.0f / 256.0f);
}

__global__ void init_hc_kernel(const float* __restrict__ meanenc,
                               const float* __restrict__ Wh, const float* __restrict__ bh,
                               const float* __restrict__ Wc, const float* __restrict__ bc,
                               float* __restrict__ cstate, bf16_t* __restrict__ xinh) {
  const int idx = blockIdx.x * blockDim.x + threadIdx.x;
  if (idx >= 64 * 512) return;
  const int b = idx >> 9, d = idx & 511;
  const float* m  = meanenc + (size_t)b * 512;
  const float* wh = Wh + (size_t)d * 512;
  const float* wc = Wc + (size_t)d * 512;
  float sh = bh[d], sc = bc[d];
  #pragma unroll 4
  for (int k = 0; k < 512; ++k) { const float mv = m[k]; sh = fmaf(mv, wh[k], sh); sc = fmaf(mv, wc[k], sc); }
  cstate[idx] = sc;
  xinh[(size_t)b * 1280 + 768 + d] = f32_to_bf16(sh);
}

// ---------------------------------------------------------------------------
extern "C" void kernel_launch(void* const* d_in, const int* in_sizes, int n_in,
                              void* d_out, int out_size, void* d_ws, size_t ws_size,
                              hipStream_t stream) {
  (void)in_sizes; (void)n_in; (void)out_size; (void)ws_size;
  const float* enc       = (const float*)d_in[0];   // [64,512,16,16]
  const int*   caps      = (const int*)  d_in[1];   // [64,32]
  const int*   lens      = (const int*)  d_in[2];   // [64]
  const float* W_enc_att = (const float*)d_in[3];
  const float* b_enc_att = (const float*)d_in[4];
  const float* W_dec_att = (const float*)d_in[5];
  const float* b_dec_att = (const float*)d_in[6];
  const float* w_full    = (const float*)d_in[7];
  const float* b_full    = (const float*)d_in[8];
  const float* emb       = (const float*)d_in[9];
  const float* W_ih      = (const float*)d_in[10];
  const float* W_hh      = (const float*)d_in[11];
  const float* b_ih      = (const float*)d_in[12];
  const float* b_hh      = (const float*)d_in[13];
  const float* W_init_h  = (const float*)d_in[14];
  const float* b_init_h  = (const float*)d_in[15];
  const float* W_init_c  = (const float*)d_in[16];
  const float* b_init_c  = (const float*)d_in[17];
  const float* W_beta    = (const float*)d_in[18];
  const float* b_beta    = (const float*)d_in[19];
  const float* W_fc      = (const float*)d_in[20];
  const float* b_fc      = (const float*)d_in[21];
  float* out = (float*)d_out;

  // workspace carve (all 256B-aligned)
  char* ws = (char*)d_ws;
  size_t off = 0;
  auto carve = [&](size_t bytes) -> void* {
    off = (off + 255) & ~(size_t)255;
    void* p = ws + off; off += bytes; return p;
  };
  float*  att1T   = (float*) carve(16384ull * 512 * 4);   // [B,ATT,P] f32
  bf16_t* encPB   = (bf16_t*)carve(16384ull * 512 * 2);   // enc [B*P, C] bf16
  bf16_t* Wfc16   = (bf16_t*)carve(16000ull * 512 * 2);
  bf16_t* Wcomb   = (bf16_t*)carve(2048ull * 1280 * 2);   // [4*DEC, EMB+ENC+DEC]
  bf16_t* Watt    = (bf16_t*)carve(1024ull * 512 * 2);    // W_dec_att | W_beta
  bf16_t* Wenc16  = (bf16_t*)carve(512ull * 512 * 2);
  bf16_t* embTB   = (bf16_t*)carve(31ull * 64 * 256 * 2);
  float*  bcomb   = (float*) carve(2048 * 4);
  float*  batt    = (float*) carve(1024 * 4);
  float*  meanenc = (float*) carve(64 * 512 * 4);
  float*  cstate  = (float*) carve(64 * 512 * 4);
  bf16_t* xinh    = (bf16_t*)carve(64ull * 1280 * 2);     // emb|awe|h per row
  float*  attg    = (float*) carve(64 * 1024 * 4);        // att2 | gate_pre
  float*  gates   = (float*) carve(64 * 2048 * 4);

  // ---- one-time setup ----
  cvt_kernel<<<512, 256, 0, stream>>>(Wfc16, W_fc, 16000 * 512);
  cvt_kernel<<<256, 256, 0, stream>>>(Wenc16, W_enc_att, 512 * 512);
  cvt_kernel<<<256, 256, 0, stream>>>(Watt, W_dec_att, 512 * 512);
  cvt_kernel<<<256, 256, 0, stream>>>(Watt + 512 * 512, W_beta, 512 * 512);
  wcomb_kernel<<<512, 256, 0, stream>>>(Wcomb, W_ih, W_hh, 2048 * 1280);
  bcomb_kernel<<<8, 256, 0, stream>>>(bcomb, b_ih, b_hh);
  batt_kernel<<<4, 256, 0, stream>>>(batt, b_dec_att, b_beta);
  embed_kernel<<<31 * 64, 256, 0, stream>>>(emb, caps, embTB);
  enc_bf16_kernel<<<2048, 256, 0, stream>>>(enc, encPB);
  mean_kernel<<<128, 256, 0, stream>>>(enc, meanenc);
  init_hc_kernel<<<128, 256, 0, stream>>>(meanenc, W_init_h, b_init_h,
                                          W_init_c, b_init_c, cstate, xinh);
  // att1T = (enc @ W_enc_att^T + b_enc_att) stored [B,ATT,P]
  gemm_att1_kernel<<<(256 * 32) / 8, 256, 0, stream>>>(encPB, Wenc16, b_enc_att, att1T);

  // ---- sequential decode ----
  const int T = 31, V = 16000;
  for (int t = 0; t < T; ++t) {
    // [att2 | gate_pre] = h @ [W_dec_att | W_beta]^T + [b_dec_att | b_beta]
    gemm_bias_kernel<<<(1 * 64) / 8, 256, 0, stream>>>(
        xinh + 768, 1280, Watt, 512, batt, attg, 1024, 64, 1024, 512);
    attention_kernel<<<64, 256, 0, stream>>>(
        att1T, attg, enc, w_full, b_full, embTB + (size_t)t * 64 * 256, xinh);
    // gates = [emb|awe|h] @ [W_ih|W_hh]^T + (b_ih + b_hh)   (M=64,N=2048,K=1280)
    gemm_bias_kernel<<<(1 * 128) / 8, 256, 0, stream>>>(
        xinh, 1280, Wcomb, 1280, bcomb, gates, 2048, 64, 2048, 1280);
    lstm_kernel<<<128, 256, 0, stream>>>(gates, cstate, xinh);
    // pred = h2 @ W_fc^T + b_fc, masked, -> out[b][t][v]  (M=64,N=16000,K=512)
    gemm_pred_kernel<<<(V / 16) / 8, 256, 0, stream>>>(
        xinh + 768, 1280, Wfc16, b_fc, lens, out, t, T, V, 512);
  }
}
